// MultiHeadedAttention_14585708937690
// MI455X (gfx1250) — matrix operand order repaired and checked by reading, not compiled, not run on hardware
//
#include <hip/hip_runtime.h>
#include <hip/hip_bf16.h>

// Problem constants (B=2, S=128, H=4, DK=128, NB=4, D=512)
#define B_  2
#define S_  128
#define H_  4
#define DK_ 128
#define D_  512
#define NB_ 4
#define BH_ (B_*H_)    // 8
#define MACT (B_*S_)   // 256
#define KCAT 1024

typedef __attribute__((ext_vector_type(16))) _Float16 v16h;
typedef __attribute__((ext_vector_type(8)))  _Float16 v8h;
typedef __attribute__((ext_vector_type(8)))  float    v8f;

// ---- CDNA5 async global->LDS copy (ASYNCcnt path), 16 bytes per lane ----
__device__ __forceinline__ void async_copy_b128(unsigned lds_off, const void* gsrc) {
  asm volatile("global_load_async_to_lds_b128 %0, %1, off"
               :: "v"(lds_off), "v"(gsrc) : "memory");
}
__device__ __forceinline__ void wait_async0() {
  asm volatile("s_wait_asynccnt 0x0" ::: "memory");
}

// ---- WMMA fragment loaders (CDNA5 16-bit operand layouts, wave32) ----
// A 16x32 f16: lane l: M=l%16, g=l/16; halves 0..7 -> K=g*8+0..7, halves 8..15 -> K=16+g*8+0..7
__device__ __forceinline__ v16h frag_a(const _Float16* __restrict__ p, int g) {
  v8h lo = *(const v8h*)(p + g * 8);
  v8h hi = *(const v8h*)(p + 16 + g * 8);
  v16h r;
#pragma unroll
  for (int t = 0; t < 8; ++t) { r[t] = lo[t]; r[8 + t] = hi[t]; }
  return r;
}
// B 32x16 f16 from [N][K] row-major operand: lane l: N=l%16, g=l/16; halves 0..7 -> K=g*8+0..7, halves 8..15 -> K=16+g*8+0..7
__device__ __forceinline__ v16h frag_b(const _Float16* __restrict__ p, int g) {
  v8h lo = *(const v8h*)(p + g * 8);
  v8h hi = *(const v8h*)(p + 16 + g * 8);
  v16h r;
#pragma unroll
  for (int t = 0; t < 8; ++t) { r[t] = lo[t]; r[8 + t] = hi[t]; }
  return r;
}

// One 16x16 f32 output tile, K-loop in steps of 32 over f16 operands.
__device__ __forceinline__ v8f tile_gemm(const _Float16* __restrict__ arow,
                                         const _Float16* __restrict__ brow,
                                         int K, int g) {
  v8f c = {};
#pragma unroll 4
  for (int k = 0; k < K; k += 32) {
    v16h a = frag_a(arow + k, g);
    v16h b = frag_b(brow + k, g);
    c = __builtin_amdgcn_wmma_f32_16x16x32_f16(false, a, false, b, (short)0, c,
                                               false, false);
  }
  return c;
}

// ---------------- pack kernels (f32 -> f16 operand staging) ----------------
__global__ void k_pack_act(const float* __restrict__ x, const float* __restrict__ bm,
                           _Float16* __restrict__ out) {
  int idx = blockIdx.x * 256 + threadIdx.x;             // 256*1024
  if (idx >= MACT * KCAT) return;
  int row = idx >> 10, col = idx & 1023;
  float v = (col < D_) ? x[row * D_ + col] : bm[row * D_ + (col - D_)];
  out[idx] = (_Float16)v;
}

// out[mat(0..2)][n=h*128+kk][kcat], weights W[3,D,H,DK]
__global__ void k_pack_w6(const float* __restrict__ Wx, const float* __restrict__ Wb,
                          _Float16* __restrict__ out) {
  int idx = blockIdx.x * 256 + threadIdx.x;             // 3*512*1024
  if (idx >= 3 * D_ * KCAT) return;
  int mat = idx / (D_ * KCAT);
  int rem = idx % (D_ * KCAT);
  int n = rem >> 10, k = rem & 1023;
  int h = n >> 7, kk = n & 127;
  float v;
  if (k < D_) v = Wx[(((size_t)mat * D_ + k) * H_ + h) * DK_ + kk];
  else        v = Wb[(((size_t)mat * D_ + (k - D_)) * H_ + h) * DK_ + kk];
  out[idx] = (_Float16)v;
}

// ph16[((b*H+h)*S+s)*DK+kk] = p_m[(b*S+s)*D + h*DK+kk]
__global__ void k_pack_p(const float* __restrict__ pm, _Float16* __restrict__ ph) {
  int idx = blockIdx.x * 256 + threadIdx.x;             // 131072
  if (idx >= BH_ * S_ * DK_) return;
  int kk = idx & 127, s = (idx >> 7) & 127, h = (idx >> 14) & 3, b = idx >> 16;
  ph[idx] = (_Float16)pm[((size_t)(b * S_ + s)) * D_ + h * DK_ + kk];
}

// bb16[(((bh)*NB+t1)*NB+t2)*DK+kk] = bb_m[((b*NB+t1)*NB+t2)*D + h*DK+kk]
__global__ void k_pack_bb(const float* __restrict__ bbm, _Float16* __restrict__ bb) {
  int idx = blockIdx.x * 256 + threadIdx.x;             // 16384
  if (idx >= BH_ * NB_ * NB_ * DK_) return;
  int kk = idx & 127, t2 = (idx >> 7) & 3, t1 = (idx >> 9) & 3;
  int h = (idx >> 11) & 3, b = idx >> 13;
  bb[idx] = (_Float16)bbm[(((size_t)(b * NB_ + t1) * NB_ + t2)) * D_ + h * DK_ + kk];
}

__global__ void k_pack_half(const float* __restrict__ s, _Float16* __restrict__ d, int n) {
  int idx = blockIdx.x * 256 + threadIdx.x;
  if (idx < n) d[idx] = (_Float16)s[idx];
}

// ---------------- GEMM kernels ----------------
// 6 projections: O[mat] = [x|b] @ W6[mat]^T ; M=256, N=512, K=1024.
// 4 waves/block share one async-staged 16-row A panel (32 KB LDS); wave w does nt=ntg*4+w.
__global__ void __launch_bounds__(128)
k_gemm_proj(const _Float16* __restrict__ Am, const _Float16* __restrict__ Ac,
            const _Float16* __restrict__ W6, float* __restrict__ qkv,
            _Float16* __restrict__ v1t, _Float16* __restrict__ v2t) {
  __shared__ _Float16 As[16 * KCAT];    // 32 KB
  int bid = blockIdx.x;                 // 6*16*8 = 768
  int ntg = bid & 7, mt = (bid >> 3) & 15, mat = bid >> 7;
  int tid = threadIdx.x;
  int wave = tid >> 5, lane = tid & 31, m = lane & 15, g = lane >> 4;
  const _Float16* A = (mat < 3) ? Am : Ac;
  const char* Ag = (const char*)(A + (size_t)mt * 16 * KCAT);   // contiguous 32 KB
  unsigned asbase = (unsigned)(uintptr_t)(&As[0]);
#pragma unroll
  for (int it = 0; it < 16; ++it) {
    int idx = it * 128 + tid;           // 2048 x 16 B
    async_copy_b128(asbase + idx * 16, Ag + (size_t)idx * 16);
  }
  wait_async0();
  __syncthreads();
  int nt = ntg * 4 + wave;              // 0..31
  const _Float16* Bt = W6 + (size_t)mat * D_ * KCAT;
  v8f c = tile_gemm(As + (size_t)m * KCAT,
                    Bt + (size_t)(nt * 16 + m) * KCAT, KCAT, g);
  float* O = qkv + (size_t)mat * BH_ * S_ * DK_;
#pragma unroll
  for (int r = 0; r < 8; ++r) {
    int row = mt * 16 + r + 8 * g;      // b*S + s
    int col = nt * 16 + m;              // h*DK + kk
    float v = c[r];
    if (mat >= 3) v = (v > 0.f) ? (v + 1.f) : __expf(v);   // elu(x)+1
    int b = row >> 7, s = row & 127, h = col >> 7, kk = col & 127;
    O[(((size_t)(b * H_ + h) * S_ + s)) * DK_ + kk] = v;
    if (mat == 2) v1t[((size_t)(b * H_ + h) * DK_ + kk) * S_ + s] = (_Float16)v;
    if (mat == 5) v2t[((size_t)(b * H_ + h) * DK_ + kk) * S_ + s] = (_Float16)v;
  }
}

// Generic small linear: out[r][n] = A[r][:] @ W[n][:] + bias[n]; K=N=128.
__global__ void k_gemm_lin(const _Float16* __restrict__ A, const _Float16* __restrict__ W,
                           const float* __restrict__ bias, float* __restrict__ out) {
  int wid = blockIdx.x;                 // Mtiles*8
  int nt = wid & 7, mt = wid >> 3;
  int lane = threadIdx.x, m = lane & 15, g = lane >> 4;
  v8f c = tile_gemm(A + (size_t)(mt * 16 + m) * DK_,
                    W + (size_t)(nt * 16 + m) * DK_, DK_, g);
#pragma unroll
  for (int r = 0; r < 8; ++r) {
    int row = mt * 16 + r + 8 * g, col = nt * 16 + m;
    out[(size_t)row * DK_ + col] = c[r] + bias[col];
  }
}

// tri_sagp fusion -> per-(row, block) f16 [mean | sqrt(cov)] (K=256) + |.|^2 sums
__global__ void k_fuse(const float* __restrict__ qkv, const float* __restrict__ pq2,
                       const float* __restrict__ pk2, const float* __restrict__ bbQ,
                       const float* __restrict__ bbK, const float* __restrict__ bb_c,
                       const float* __restrict__ p_c, const int* __restrict__ b_seq,
                       _Float16* __restrict__ A16, _Float16* __restrict__ B16,
                       float* __restrict__ sumA, float* __restrict__ sumB) {
  int idx = blockIdx.x;                 // 2*8*128*4 = 8192
  int side = idx >> 12;                 // 0 = Q(rows i), 1 = K(cols j)
  int rem = idx & 4095;
  int bh = rem >> 9, i = (rem >> 2) & 127, t = rem & 3;
  int b = bh >> 2, h = bh & 3;
  int k = threadIdx.x;                  // 0..127
  int sI = b_seq[b * S_ + i];
  const size_t NHD = (size_t)BH_ * S_ * DK_;
  const float* m1p = qkv + (side ? 1 : 0) * NHD;      // q1 / k1
  const float* c1p = qkv + (side ? 4 : 3) * NHD;      // q2 / k2
  const float* m3p = side ? pk2 : pq2;
  const float* bbL = side ? bbK : bbQ;
  size_t ro = ((size_t)bh * S_ + i) * DK_ + k;
  float m1 = m1p[ro], c1 = c1p[ro], m3 = m3p[ro];
  float c3 = p_c[((size_t)(b * S_ + i)) * D_ + h * DK_ + k];
  int t1 = side ? sI : t;               // sj index into bb
  int t2 = side ? t : sI;               // si index into bb
  float m2 = bbL[(((size_t)bh * NB_ + t1) * NB_ + t2) * DK_ + k];
  float c2 = bb_c[(((size_t)(b * NB_ + t1) * NB_ + t2)) * D_ + h * DK_ + k];
  const float eps = 1e-24f;
  float p1 = 1.f / fmaxf(c1, eps), p2 = 1.f / fmaxf(c2, eps), p3 = 1.f / fmaxf(c3, eps);
  float cc = 1.f / (p1 + p2 + p3);
  float fm = cc * (m1 * p1 + m2 * p2 + m3 * p3);
  float fs = sqrtf(fmaxf(cc, eps));
  _Float16* dst = (side ? B16 : A16) + (((size_t)bh * NB_ + t) * S_ + i) * 256;
  dst[k] = (_Float16)fm;
  dst[128 + k] = (_Float16)fs;
  __shared__ float red[128];
  red[k] = fm * fm + fs * fs;
  __syncthreads();
#pragma unroll
  for (int st = 64; st > 0; st >>= 1) {
    if (k < st) red[k] += red[k + st];
    __syncthreads();
  }
  if (k == 0) (side ? sumB : sumA)[((size_t)bh * S_ + i) * NB_ + t] = red[0];
}

// Cross-term GEMMs: G[bh][t][u] = A16[bh,t] (128x256) @ B16[bh,u]^T -> 128x128
// 4 waves/block share one async-staged 16-row A panel (8 KB LDS).
__global__ void __launch_bounds__(128)
k_gemm_G(const _Float16* __restrict__ A16, const _Float16* __restrict__ B16,
         float* __restrict__ G) {
  __shared__ _Float16 As[16 * 256];     // 8 KB
  int bid = blockIdx.x;                 // 8*4*4*8*2 = 2048
  int ntg = bid & 1, mtb = (bid >> 1) & 7;
  int u = (bid >> 4) & 3, t = (bid >> 6) & 3, bh = bid >> 8;
  int tid = threadIdx.x;
  int wave = tid >> 5, lane = tid & 31, m = lane & 15, g = lane >> 4;
  const char* Ag = (const char*)(A16 + (((size_t)bh * NB_ + t) * S_ + mtb * 16) * 256);
  unsigned asbase = (unsigned)(uintptr_t)(&As[0]);
#pragma unroll
  for (int it = 0; it < 4; ++it) {
    int idx = it * 128 + tid;           // 512 x 16 B
    async_copy_b128(asbase + idx * 16, Ag + (size_t)idx * 16);
  }
  wait_async0();
  __syncthreads();
  int ntb = ntg * 4 + wave;             // 0..7
  const _Float16* Bt = B16 + ((size_t)bh * NB_ + u) * S_ * 256;
  v8f c = tile_gemm(As + (size_t)m * 256,
                    Bt + (size_t)(ntb * 16 + m) * 256, 256, g);
  float* O = G + ((size_t)bh * 16 + t * 4 + u) * S_ * S_;
#pragma unroll
  for (int r = 0; r < 8; ++r)
    O[(size_t)(mtb * 16 + r + 8 * g) * S_ + ntb * 16 + m] = c[r];
}

// scores + softmax (one block per attention row)
__global__ void k_scores(const float* __restrict__ G, const float* __restrict__ sumA,
                         const float* __restrict__ sumB, const int* __restrict__ b_seq,
                         float* __restrict__ probsOut, _Float16* __restrict__ probs16,
                         _Float16* __restrict__ p2_16) {
  int bi = blockIdx.x;                  // bh*128 + i
  int bh = bi >> 7, i = bi & 127, b = bh >> 2;
  int j = threadIdx.x;
  int si = b_seq[b * S_ + i], sj = b_seq[b * S_ + j];
  float gdot = G[(((size_t)bh * 16 + sj * 4 + si) * S_ + i) * S_ + j];
  float w2 = sumA[((size_t)bh * S_ + i) * NB_ + sj] +
             sumB[((size_t)bh * S_ + j) * NB_ + si] - 2.f * gdot;
  float score = -w2 * 0.08838834764831845f;   // 1/sqrt(DK)
  __shared__ float red[128];
  red[j] = score;
  __syncthreads();
#pragma unroll
  for (int st = 64; st > 0; st >>= 1) {
    if (j < st) red[j] = fmaxf(red[j], red[j + st]);
    __syncthreads();
  }
  float mx = red[0];
  __syncthreads();
  float e = __expf(score - mx);
  red[j] = e;
  __syncthreads();
#pragma unroll
  for (int st = 64; st > 0; st >>= 1) {
    if (j < st) red[j] += red[j + st];
    __syncthreads();
  }
  float p = e / red[0];
  probsOut[(size_t)bi * S_ + j] = p;
  probs16[(size_t)bi * S_ + j] = (_Float16)p;
  p2_16[(size_t)bi * S_ + j]   = (_Float16)(p * p);
}

// probs @ v1 and probs^2 @ v2 -> f16 ctx in [B,S,D] layout
__global__ void k_gemm_av(const _Float16* __restrict__ probs16, const _Float16* __restrict__ p2_16,
                          const _Float16* __restrict__ v1t, const _Float16* __restrict__ v2t,
                          _Float16* __restrict__ mc16, _Float16* __restrict__ cc16) {
  int wid = blockIdx.x;                 // 2*8*8*8 = 1024
  int nt = wid & 7, mt = (wid >> 3) & 7, bh = (wid >> 6) & 7, which = wid >> 9;
  int lane = threadIdx.x, m = lane & 15, g = lane >> 4;
  const _Float16* A  = (which ? p2_16 : probs16) + (size_t)bh * S_ * S_;
  const _Float16* Bt = (which ? v2t : v1t) + (size_t)bh * DK_ * S_;
  v8f c = tile_gemm(A + (size_t)(mt * 16 + m) * S_,
                    Bt + (size_t)(nt * 16 + m) * S_, S_, g);
  _Float16* O = which ? cc16 : mc16;
  int b = bh >> 2, h = bh & 3;
#pragma unroll
  for (int r = 0; r < 8; ++r) {
    int row = mt * 16 + r + 8 * g;      // s
    int col = nt * 16 + m;              // kk
    O[((size_t)(b * S_ + row)) * D_ + h * DK_ + col] = (_Float16)c[r];
  }
}

// ctx @ W^T + bias + residual -> pre-LN f32; async-staged A panel (16 KB LDS)
__global__ void __launch_bounds__(128)
k_gemm_out(const _Float16* __restrict__ mc16, const _Float16* __restrict__ cc16,
           const _Float16* __restrict__ Wout, const float* __restrict__ mean_b,
           const float* __restrict__ cov_b, const float* __restrict__ x_m,
           const float* __restrict__ x_c, float* __restrict__ pre) {
  __shared__ _Float16 As[16 * D_];      // 16 KB
  int bid = blockIdx.x;                 // 2*16*8 = 256
  int ntg = bid & 7, mt = (bid >> 3) & 15, which = bid >> 7;
  int tid = threadIdx.x;
  int wave = tid >> 5, lane = tid & 31, m = lane & 15, g = lane >> 4;
  const _Float16* A = which ? cc16 : mc16;
  const char* Ag = (const char*)(A + (size_t)mt * 16 * D_);     // contiguous 16 KB
  unsigned asbase = (unsigned)(uintptr_t)(&As[0]);
#pragma unroll
  for (int it = 0; it < 8; ++it) {
    int idx = it * 128 + tid;           // 1024 x 16 B
    async_copy_b128(asbase + idx * 16, Ag + (size_t)idx * 16);
  }
  wait_async0();
  __syncthreads();
  int nt = ntg * 4 + wave;              // 0..31
  const _Float16* Bt = Wout + (size_t)which * D_ * D_;
  const float* bias  = which ? cov_b : mean_b;
  const float* resid = which ? x_c : x_m;
  float* O = pre + (size_t)which * MACT * D_;
  v8f c = tile_gemm(As + (size_t)m * D_,
                    Bt + (size_t)(nt * 16 + m) * D_, D_, g);
#pragma unroll
  for (int r = 0; r < 8; ++r) {
    int row = mt * 16 + r + 8 * g, col = nt * 16 + m;
    O[(size_t)row * D_ + col] = c[r] + bias[col] + resid[(size_t)row * D_ + col];
  }
}

// LayerNorm rows of 512; pre rows 0..255 -> mean_h, 256..511 -> cov_h
__global__ void k_ln(const float* __restrict__ pre, const float* __restrict__ ln_w,
                     const float* __restrict__ ln_b, float* __restrict__ out) {
  int rowid = blockIdx.x;               // 512
  const float* x = pre + (size_t)rowid * D_;
  float* o = out + (size_t)rowid * D_;
  int tid = threadIdx.x;                // 128
  float vals[4], s = 0.f;
#pragma unroll
  for (int q = 0; q < 4; ++q) { vals[q] = x[tid + q * 128]; s += vals[q]; }
  __shared__ float red[128];
  red[tid] = s;
  __syncthreads();
#pragma unroll
  for (int st = 64; st > 0; st >>= 1) {
    if (tid < st) red[tid] += red[tid + st];
    __syncthreads();
  }
  float mean = red[0] / (float)D_;
  __syncthreads();
  float vs = 0.f;
#pragma unroll
  for (int q = 0; q < 4; ++q) { float d = vals[q] - mean; vs += d * d; }
  red[tid] = vs;
  __syncthreads();
#pragma unroll
  for (int st = 64; st > 0; st >>= 1) {
    if (tid < st) red[tid] += red[tid + st];
    __syncthreads();
  }
  float inv = rsqrtf(red[0] / (float)D_ + 1e-12f);
#pragma unroll
  for (int q = 0; q < 4; ++q) {
    int col = tid + q * 128;
    o[col] = (vals[q] - mean) * inv * ln_w[col] + ln_b[col];
  }
}

// ---------------- launch ----------------
extern "C" void kernel_launch(void* const* d_in, const int* in_sizes, int n_in,
                              void* d_out, int out_size, void* d_ws, size_t ws_size,
                              hipStream_t stream) {
  const float* x_m  = (const float*)d_in[0];
  const float* x_c  = (const float*)d_in[1];
  const float* b_m  = (const float*)d_in[2];
  const float* b_c  = (const float*)d_in[3];
  const float* bb_m = (const float*)d_in[4];
  const float* bb_c = (const float*)d_in[5];
  const float* p_m  = (const float*)d_in[6];
  const float* p_c  = (const float*)d_in[7];
  const float* W_xm = (const float*)d_in[8];
  const float* W_xc = (const float*)d_in[9];
  const float* W_bm = (const float*)d_in[10];
  const float* W_bc = (const float*)d_in[11];
  const float* Wq1_w = (const float*)d_in[12];
  const float* Wq1_b = (const float*)d_in[13];
  const float* Wq2_w = (const float*)d_in[14];
  const float* Wq2_b = (const float*)d_in[15];
  const float* Wk1_w = (const float*)d_in[16];
  const float* Wk1_b = (const float*)d_in[17];
  const float* Wk2_w = (const float*)d_in[18];
  const float* Wk2_b = (const float*)d_in[19];
  const float* mean_w = (const float*)d_in[20];
  const float* mean_b = (const float*)d_in[21];
  const float* cov_w  = (const float*)d_in[22];
  const float* cov_b  = (const float*)d_in[23];
  const float* ln_w   = (const float*)d_in[24];
  const float* ln_b   = (const float*)d_in[25];
  const int*   b_seq  = (const int*)d_in[26];

  float* outF = (float*)d_out;
  float* probsOut = outF + 2 * MACT * D_;   // after mean_h, cov_h

  char* w = (char*)d_ws;
  auto alloc = [&](size_t bytes) -> void* {
    void* p = (void*)w;
    w += (bytes + 255) & ~(size_t)255;
    return p;
  };
  _Float16* Am16 = (_Float16*)alloc((size_t)MACT * KCAT * 2);
  _Float16* Ac16 = (_Float16*)alloc((size_t)MACT * KCAT * 2);
  _Float16* W6   = (_Float16*)alloc((size_t)6 * D_ * KCAT * 2);
  _Float16* ph16 = (_Float16*)alloc((size_t)BH_ * S_ * DK_ * 2);
  _Float16* bb16 = (_Float16*)alloc((size_t)BH_ * NB_ * NB_ * DK_ * 2);
  _Float16* W4   = (_Float16*)alloc((size_t)4 * DK_ * DK_ * 2);
  _Float16* Wout = (_Float16*)alloc((size_t)2 * D_ * D_ * 2);
  _Float16* A16  = (_Float16*)alloc((size_t)BH_ * NB_ * S_ * 256 * 2);
  _Float16* B16  = (_Float16*)alloc((size_t)BH_ * NB_ * S_ * 256 * 2);
  _Float16* probs16 = (_Float16*)alloc((size_t)BH_ * S_ * S_ * 2);
  _Float16* p2_16   = (_Float16*)alloc((size_t)BH_ * S_ * S_ * 2);
  _Float16* v1t  = (_Float16*)alloc((size_t)BH_ * DK_ * S_ * 2);
  _Float16* v2t  = (_Float16*)alloc((size_t)BH_ * DK_ * S_ * 2);
  _Float16* mc16 = (_Float16*)alloc((size_t)MACT * D_ * 2);
  _Float16* cc16 = (_Float16*)alloc((size_t)MACT * D_ * 2);
  float* qkv  = (float*)alloc((size_t)6 * BH_ * S_ * DK_ * 4);
  float* pq2  = (float*)alloc((size_t)BH_ * S_ * DK_ * 4);
  float* pk2  = (float*)alloc((size_t)BH_ * S_ * DK_ * 4);
  float* bbQ  = (float*)alloc((size_t)BH_ * NB_ * NB_ * DK_ * 4);
  float* bbK  = (float*)alloc((size_t)BH_ * NB_ * NB_ * DK_ * 4);
  float* sumA = (float*)alloc((size_t)BH_ * S_ * NB_ * 4);
  float* sumB = (float*)alloc((size_t)BH_ * S_ * NB_ * 4);
  float* G    = (float*)alloc((size_t)BH_ * 16 * S_ * S_ * 4);
  float* pre  = (float*)alloc((size_t)2 * MACT * D_ * 4);
  (void)ws_size; (void)n_in; (void)in_sizes; (void)out_size;

  // ---- stage f16 operands ----
  k_pack_act<<<1024, 256, 0, stream>>>(x_m, b_m, Am16);
  k_pack_act<<<1024, 256, 0, stream>>>(x_c, b_c, Ac16);
  k_pack_w6<<<6144, 256, 0, stream>>>(W_xm, W_bm, W6);
  k_pack_w6<<<6144, 256, 0, stream>>>(W_xc, W_bc, W6 + (size_t)3 * D_ * KCAT);
  k_pack_p<<<512, 256, 0, stream>>>(p_m, ph16);
  k_pack_bb<<<64, 256, 0, stream>>>(bb_m, bb16);
  k_pack_half<<<64, 256, 0, stream>>>(Wq1_w, W4 + 0 * DK_ * DK_, DK_ * DK_);
  k_pack_half<<<64, 256, 0, stream>>>(Wq2_w, W4 + 1 * DK_ * DK_, DK_ * DK_);
  k_pack_half<<<64, 256, 0, stream>>>(Wk1_w, W4 + 2 * DK_ * DK_, DK_ * DK_);
  k_pack_half<<<64, 256, 0, stream>>>(Wk2_w, W4 + 3 * DK_ * DK_, DK_ * DK_);
  k_pack_half<<<1024, 256, 0, stream>>>(mean_w, Wout, D_ * D_);
  k_pack_half<<<1024, 256, 0, stream>>>(cov_w, Wout + (size_t)D_ * D_, D_ * D_);

  // ---- QKV projections (WMMA + async LDS staging) ----
  k_gemm_proj<<<768, 128, 0, stream>>>(Am16, Ac16, W6, qkv, v1t, v2t);

  // ---- small linears (WMMA): pq2, pk2, bbQ, bbK ----
  k_gemm_lin<<<(BH_ * S_ / 16) * 8, 32, 0, stream>>>(ph16, W4 + 1 * DK_ * DK_, Wq2_b, pq2);
  k_gemm_lin<<<(BH_ * S_ / 16) * 8, 32, 0, stream>>>(ph16, W4 + 3 * DK_ * DK_, Wk2_b, pk2);
  k_gemm_lin<<<(BH_ * NB_ * NB_ / 16) * 8, 32, 0, stream>>>(bb16, W4 + 0 * DK_ * DK_, Wq1_b, bbQ);
  k_gemm_lin<<<(BH_ * NB_ * NB_ / 16) * 8, 32, 0, stream>>>(bb16, W4 + 2 * DK_ * DK_, Wk1_b, bbK);

  // ---- tri_sagp fusion ----
  k_fuse<<<8192, 128, 0, stream>>>(qkv, pq2, pk2, bbQ, bbK, bb_c, p_c, b_seq,
                                   A16, B16, sumA, sumB);

  // ---- cross-term block GEMMs (WMMA + async LDS staging) ----
  k_gemm_G<<<2048, 128, 0, stream>>>(A16, B16, G);

  // ---- scores + softmax ----
  k_scores<<<1024, 128, 0, stream>>>(G, sumA, sumB, b_seq, probsOut, probs16, p2_16);

  // ---- attention-value GEMMs (WMMA) ----
  k_gemm_av<<<1024, 32, 0, stream>>>(probs16, p2_16, v1t, v2t, mc16, cc16);

  // ---- output linear + residual (WMMA + async LDS staging) ----
  k_gemm_out<<<256, 128, 0, stream>>>(mc16, cc16, Wout, mean_b, cov_b, x_m, x_c, pre);

  // ---- LayerNorm -> d_out ----
  k_ln<<<512, 128, 0, stream>>>(pre, ln_w, ln_b, outF);
}
